// MambaBlock_69398081569202
// MI455X (gfx1250) — compile-verified, matrix operand order checked by reading, not run on hardware
//
#include <hip/hip_runtime.h>
#include <hip/hip_bf16.h>
#include <math.h>

typedef _Float16 f16;
typedef __attribute__((ext_vector_type(16))) _Float16 v16h;
typedef __attribute__((ext_vector_type(8)))  _Float16 v8h;
typedef __attribute__((ext_vector_type(8)))  float    v8f;

#define BQ     8
#define CCH    256
#define LL     4096
#define TOK    (BQ * LL)       // 32768 tokens
#define DIN    512             // D_INNER
#define DSTATE 16
#define CONVD  544             // D_INNER + 2*D_STATE
#define NHEAD  8
#define HD     64
#define NPJ    1064            // D_IN_PROJ
#define ZLD    1088            // padded to 17 * 64 (block N-tile width)

// --- gfx1250 async global->LDS path (ASYNCcnt), with portable fallback ------
#if defined(__has_builtin)
# if __has_builtin(__builtin_amdgcn_global_load_async_to_lds_b128) && \
     __has_builtin(__builtin_amdgcn_s_wait_asynccnt)
#  define ASYNC_LDS 1
# endif
#endif
#ifndef ASYNC_LDS
# define ASYNC_LDS 0
#endif

#if ASYNC_LDS
// Builtin parameter type per hipcc diagnostic: pointer to 16B int vector,
// global (AS1) source and LDS (AS3) destination.
typedef int v4i_ __attribute__((vector_size(16)));
typedef __attribute__((address_space(1))) v4i_* as1_v4i_p;
typedef __attribute__((address_space(3))) v4i_* as3_v4i_p;
#endif

// ---------------------------------------------------------------------------
// Weight pre-pack: f32 -> f16, transposed to [N][K] so B-fragments are
// K-contiguous (2 x b128 per lane).
// ---------------------------------------------------------------------------
__global__ void k_pack_win(const float* __restrict__ Win, f16* __restrict__ WinT) {
    int idx = blockIdx.x * 256 + threadIdx.x;
    if (idx >= ZLD * CCH) return;
    int n = idx / CCH, k = idx % CCH;
    float v = (n < NPJ) ? Win[(size_t)k * NPJ + n] : 0.f;
    WinT[(size_t)n * CCH + k] = (f16)v;
}

__global__ void k_pack_wout(const float* __restrict__ Wout, f16* __restrict__ WoutT) {
    int idx = blockIdx.x * 256 + threadIdx.x;
    if (idx >= CCH * DIN) return;
    int c = idx / DIN, d = idx % DIN;
    WoutT[(size_t)c * DIN + d] = (f16)Wout[(size_t)d * CCH + c];
}

// ---------------------------------------------------------------------------
// LayerNorm over C=256 per token; one wave per token (8 channels per lane).
// ---------------------------------------------------------------------------
__global__ __launch_bounds__(256) void k_layernorm(const float* __restrict__ x,
                                                   const float* __restrict__ g,
                                                   const float* __restrict__ beta,
                                                   f16* __restrict__ unh) {
    int wave = threadIdx.x >> 5;
    int lane = threadIdx.x & 31;
    int t = blockIdx.x * 8 + wave;
    int b = t >> 12, l = t & (LL - 1);
    const float* xp = x + (size_t)b * CCH * LL + l;
    int c0 = lane * 8;
    float v[8];
    float s = 0.f;
#pragma unroll
    for (int i = 0; i < 8; ++i) { v[i] = xp[(size_t)(c0 + i) * LL]; s += v[i]; }
#pragma unroll
    for (int off = 16; off; off >>= 1) s += __shfl_xor(s, off, 32);
    float mu = s * (1.f / 256.f);
    float q = 0.f;
#pragma unroll
    for (int i = 0; i < 8; ++i) { float d = v[i] - mu; q += d * d; }
#pragma unroll
    for (int off = 16; off; off >>= 1) q += __shfl_xor(q, off, 32);
    float rstd = rsqrtf(q * (1.f / 256.f) + 1e-5f);
    v8h ov;
#pragma unroll
    for (int i = 0; i < 8; ++i)
        ov[i] = (f16)((v[i] - mu) * rstd * g[c0 + i] + beta[c0 + i]);
    *(v8h*)(unh + (size_t)t * CCH + c0) = ov;
}

// ---------------------------------------------------------------------------
// Fragment loaders: 16x32 f16 tile (CDNA5 ISA 7.12.2 A layout).
// lanes 0-15 : row = lane,    halves = K {0..7} U {16..23}
// lanes 16-31: row = lane-16, halves = K {8..15} U {24..31}
// ---------------------------------------------------------------------------
__device__ inline v16h load_frag(const f16* __restrict__ A, int lda, int row,
                                 int k0, int lane) {
    int kb = (lane & 16) ? 8 : 0;
    const f16* p = A + (size_t)row * lda + k0 + kb;
    union { v16h v; v8h h[2]; } u;
    u.h[0] = *(const v8h*)p;
    u.h[1] = *(const v8h*)(p + 16);
    return u.v;
}

__device__ inline v16h load_frag_lds(const f16* S, int ldr, int row, int k0, int lane) {
    int kb = (lane & 16) ? 8 : 0;
    const f16* p = S + row * ldr + k0 + kb;
    union { v16h v; v8h h[2]; } u;
    u.h[0] = *(const v8h*)p;
    u.h[1] = *(const v8h*)(p + 16);
    return u.v;
}

// ---------------------------------------------------------------------------
// Register-blocked GEMM: block = 4 waves = 128 rows x 64 cols.
// Each wave: 2 M-tiles x 4 N-tiles (8 accumulators), A frag reused 4x.
// B panel (64 x 256 halves) staged into LDS via async global->LDS (ASYNCcnt),
// read back as conflict-free ds_load_b128 (row stride padded by 8 halves).
// FUSE=false: store f32 C with leading dim ZLD (GEMM1).
// FUSE=true : out[b,c,l] = x[b,c,l] + acc (GEMM2, fused transpose+residual).
// ---------------------------------------------------------------------------
template<int KD, bool FUSE>
__global__ __launch_bounds__(128) void k_gemm(const f16* __restrict__ A,
                                              const f16* __restrict__ BT,
                                              float* __restrict__ Cout,
                                              const float* __restrict__ xin,
                                              float* __restrict__ outT) {
    constexpr int KSTAGE = 256;
    constexpr int LROW = KSTAGE + 8;          // padded halves per LDS row (528 B)
    __shared__ f16 bsm[64 * LROW];

    int lane = threadIdx.x & 31;
    int wv = threadIdx.x >> 5;
    int tm = blockIdx.x * 128 + wv * 32;
    int tn0 = blockIdx.y * 64;

    v8f zf = {};
    v8f acc[8];
#pragma unroll
    for (int i = 0; i < 8; ++i) acc[i] = zf;

    for (int ks = 0; ks < KD; ks += KSTAGE) {
        // ---- stage B panel: 64 rows x 256 halves = 2048 b128 chunks / 16 per thread
        for (int c = threadIdx.x; c < 64 * (KSTAGE / 8); c += 128) {
            int row = c >> 5;                 // chunk / 32
            int kc = (c & 31) << 3;           // (chunk % 32) * 8 halves
            const f16* gp = BT + (size_t)(tn0 + row) * KD + ks + kc;
            f16* lp = bsm + row * LROW + kc;
#if ASYNC_LDS
            __builtin_amdgcn_global_load_async_to_lds_b128(
                (as1_v4i_p)gp, (as3_v4i_p)lp, 0, 0);
#else
            *(v8h*)lp = *(const v8h*)gp;
#endif
        }
#if ASYNC_LDS
        __builtin_amdgcn_s_wait_asynccnt(0);
#endif
        __syncthreads();

        int rA0 = tm + (lane & 15);
#pragma unroll
        for (int k0 = 0; k0 < KSTAGE; k0 += 32) {
            v16h a0 = load_frag(A, KD, rA0, ks + k0, lane);
            v16h a1 = load_frag(A, KD, rA0 + 16, ks + k0, lane);
#pragma unroll
            for (int j = 0; j < 4; ++j) {
                v16h bj = load_frag_lds(bsm, LROW, j * 16 + (lane & 15), k0, lane);
                acc[j * 2 + 0] = __builtin_amdgcn_wmma_f32_16x16x32_f16(
                    false, a0, false, bj, (short)0, acc[j * 2 + 0], false, false);
                acc[j * 2 + 1] = __builtin_amdgcn_wmma_f32_16x16x32_f16(
                    false, a1, false, bj, (short)0, acc[j * 2 + 1], false, false);
            }
        }
        __syncthreads();
    }

    // ---- epilogue: C/D layout = VGPR r -> M = r + (lane>=16 ? 8 : 0), N = lane&15
    int mbase = tm + ((lane >> 4) << 3);
    int colb = tn0 + (lane & 15);
#pragma unroll
    for (int j = 0; j < 4; ++j) {
        int col = colb + j * 16;
#pragma unroll
        for (int i = 0; i < 2; ++i) {
            v8f a = acc[j * 2 + i];
#pragma unroll
            for (int r = 0; r < 8; ++r) {
                int m = mbase + i * 16 + r;
                if constexpr (FUSE) {
                    int bq = m >> 12, l = m & (LL - 1);
                    size_t oi = (size_t)bq * CCH * LL + (size_t)col * LL + l;
                    outT[oi] = xin[oi] + a[r];
                } else {
                    Cout[(size_t)m * ZLD + col] = a[r];
                }
            }
        }
    }
}

// ---------------------------------------------------------------------------
// Depthwise causal conv (width 4) + SiLU over the xBC slice of zxbcdt.
// ---------------------------------------------------------------------------
__global__ __launch_bounds__(256) void k_conv(const float* __restrict__ zx,
                                              const float* __restrict__ cw,
                                              const float* __restrict__ cb,
                                              float* __restrict__ xbca) {
    size_t idx = (size_t)blockIdx.x * 256 + threadIdx.x;
    if (idx >= (size_t)TOK * CONVD) return;
    int ch = (int)(idx % CONVD);
    size_t t = idx / CONVD;
    int l = (int)(t & (LL - 1));
    float acc = cb[ch];
#pragma unroll
    for (int k = 0; k < 4; ++k) {
        int ls = l - 3 + k;
        if (ls >= 0)
            acc += zx[(t - 3 + k) * ZLD + DIN + ch] * cw[ch * 4 + k];
    }
    xbca[t * CONVD + ch] = acc / (1.f + expf(-acc));  // SiLU
}

// ---------------------------------------------------------------------------
// Selective scan: B x NHEAD x (4 p-blocks) = 256 blocks, 256 thr = 16p x 16n.
// ---------------------------------------------------------------------------
__global__ __launch_bounds__(256) void k_scan(const float* __restrict__ zx,
                                              const float* __restrict__ xbca,
                                              const float* __restrict__ dt_bias,
                                              const float* __restrict__ A_log,
                                              const float* __restrict__ Dp,
                                              float* __restrict__ ys) {
    int pblk = blockIdx.x & 3;
    int h = (blockIdx.x >> 2) & 7;
    int b = blockIdx.x >> 5;
    int pl = threadIdx.x >> 4;
    int n = threadIdx.x & 15;
    int p = pblk * 16 + pl;
    float Ah = -expf(A_log[h]);
    float dtb = dt_bias[h];
    float Dh = Dp[h];
    float hs = 0.f;
    for (int l = 0; l < LL; ++l) {
        size_t t = (size_t)b * LL + l;
        const float* xb = xbca + t * CONVD;
        float dtraw = zx[t * ZLD + (DIN + CONVD) + h] + dtb;
        float dt = (dtraw > 20.f) ? dtraw : log1pf(expf(dtraw));  // softplus
        float dA = expf(dt * Ah);
        float xv = xb[h * HD + p];
        float Bv = xb[DIN + n];
        float Cv = xb[DIN + DSTATE + n];
        hs = dA * hs + (dt * Bv) * xv;
        float yp = hs * Cv;
#pragma unroll
        for (int off = 8; off; off >>= 1) yp += __shfl_xor(yp, off, 32);
        if (n == 0) ys[t * DIN + h * HD + p] = yp + Dh * xv;
        if (l + 1 < LL)  // prefetch next token row (global_prefetch_b8)
            __builtin_prefetch(xb + CONVD + h * HD + p, 0, 1);
    }
}

// ---------------------------------------------------------------------------
// Gate with SiLU(z), RMSNorm over 512, cast to f16 for GEMM2.
// ---------------------------------------------------------------------------
__global__ __launch_bounds__(256) void k_gate_rms(const float* __restrict__ ys,
                                                  const float* __restrict__ zx,
                                                  const float* __restrict__ rms_w,
                                                  f16* __restrict__ yh) {
    __shared__ float red[8];
    size_t t = blockIdx.x;
    int tid = threadIdx.x;
    float gv[2];
    float ss = 0.f;
#pragma unroll
    for (int i = 0; i < 2; ++i) {
        int d = tid + i * 256;
        float y = ys[t * DIN + d];
        float z = zx[t * ZLD + d];
        float g = y * (z / (1.f + expf(-z)));
        gv[i] = g;
        ss += g * g;
    }
#pragma unroll
    for (int off = 16; off; off >>= 1) ss += __shfl_xor(ss, off, 32);
    if ((tid & 31) == 0) red[tid >> 5] = ss;
    __syncthreads();
    float tot = red[0] + red[1] + red[2] + red[3] + red[4] + red[5] + red[6] + red[7];
    float rstd = rsqrtf(tot * (1.f / 512.f) + 1e-5f);
#pragma unroll
    for (int i = 0; i < 2; ++i) {
        int d = tid + i * 256;
        yh[t * DIN + d] = (f16)(gv[i] * rstd * rms_w[d]);
    }
}

// ---------------------------------------------------------------------------
extern "C" void kernel_launch(void* const* d_in, const int* in_sizes, int n_in,
                              void* d_out, int out_size, void* d_ws, size_t ws_size,
                              hipStream_t stream) {
    const float* x       = (const float*)d_in[0];
    const float* ln_g    = (const float*)d_in[1];
    const float* ln_b    = (const float*)d_in[2];
    const float* Win     = (const float*)d_in[3];
    const float* Wout    = (const float*)d_in[4];
    const float* conv_w  = (const float*)d_in[5];
    const float* conv_b  = (const float*)d_in[6];
    const float* dt_bias = (const float*)d_in[7];
    const float* A_log   = (const float*)d_in[8];
    const float* Dv      = (const float*)d_in[9];
    const float* rms_w   = (const float*)d_in[10];

    char* ws = (char*)d_ws;
    size_t off = 0;
    auto alloc = [&](size_t bytes) {
        size_t o = off;
        off += (bytes + 255) & ~(size_t)255;
        return o;
    };
    f16*   WinT  = (f16*)(ws + alloc((size_t)ZLD * CCH * sizeof(f16)));
    f16*   WoutT = (f16*)(ws + alloc((size_t)CCH * DIN * sizeof(f16)));
    f16*   unh   = (f16*)(ws + alloc((size_t)TOK * CCH * sizeof(f16)));
    float* zx    = (float*)(ws + alloc((size_t)TOK * ZLD * sizeof(float)));
    float* xbca  = (float*)(ws + alloc((size_t)TOK * CONVD * sizeof(float)));
    float* ys    = (float*)(ws + alloc((size_t)TOK * DIN * sizeof(float)));
    f16*   yh    = (f16*)(ws + alloc((size_t)TOK * DIN * sizeof(f16)));

    k_pack_win <<<(ZLD * CCH + 255) / 256, 256, 0, stream>>>(Win, WinT);
    k_pack_wout<<<(CCH * DIN + 255) / 256, 256, 0, stream>>>(Wout, WoutT);
    k_layernorm<<<TOK / 8, 256, 0, stream>>>(x, ln_g, ln_b, unh);
    k_gemm<CCH, false><<<dim3(TOK / 128, ZLD / 64), 128, 0, stream>>>(
        unh, WinT, zx, nullptr, nullptr);
    k_conv     <<<(int)(((size_t)TOK * CONVD + 255) / 256), 256, 0, stream>>>(zx, conv_w, conv_b, xbca);
    k_scan     <<<256, 256, 0, stream>>>(zx, xbca, dt_bias, A_log, Dv, ys);
    k_gate_rms <<<TOK, 256, 0, stream>>>(ys, zx, rms_w, yh);
    k_gemm<DIN, true><<<dim3(TOK / 128, CCH / 64), 128, 0, stream>>>(
        yh, WoutT, nullptr, x, (float*)d_out);
}
